// DevignModel_60275571032228
// MI455X (gfx1250) — compile-verified, matrix operand order checked by reading, not run on hardware
//
#include <hip/hip_runtime.h>
#include <hip/hip_bf16.h>

// ---------------- problem constants ----------------
#define NN      32768      // total nodes = B * N_NODES
#define EDGES   393216
#define BGR     128        // graphs
#define ODIM    256
#define IDIM    128
#define CDIM    384

typedef __attribute__((ext_vector_type(16))) __bf16 v16bf;
typedef __attribute__((ext_vector_type(8)))  float  v8f;

// ---------------- WMMA helpers ----------------
__device__ __forceinline__ v8f wmma_bf16(v16bf a, v16bf b, v8f c) {
  // D = A(16x32 bf16) x B(32x16 bf16) + C(16x16 f32)
  return __builtin_amdgcn_wmma_f32_16x16x32_bf16(false, a, false, b, (short)0, c,
                                                 false, false);
}

// A operand: f32 row in memory -> bf16 regs in ISA layout.
// lane l<16 holds K = kb+0..7 and kb+16..23 ; lane>=16 holds +8 shifted halves.
__device__ __forceinline__ v16bf load_a_f32(const float* __restrict__ row, int kb, int lane) {
  const float* p = row + kb + ((lane >> 4) << 3);
  v16bf v;
#pragma unroll
  for (int i = 0; i < 8; ++i) {
    v[i]     = (__bf16)p[i];
    v[i + 8] = (__bf16)p[i + 16];
  }
  return v;
}

// B operand: pre-transposed bf16 weights [N][K]; per-lane row = (n)*K.
__device__ __forceinline__ v16bf load_b_bf16(const __bf16* __restrict__ row, int kb, int lane) {
  const __bf16* p = row + kb + ((lane >> 4) << 3);
  v16bf v;
#pragma unroll
  for (int i = 0; i < 8; ++i) {
    v[i]     = p[i];
    v[i + 8] = p[i + 16];
  }
  return v;
}

__device__ __forceinline__ float sigm(float x) { return 1.0f / (1.0f + expf(-x)); }

// ---------------- weight prep ----------------
// Wt[n*K + k] = W[k*N + n]   (f32 [K][N] -> bf16 [N][K])
__global__ void prep_t_kernel(const float* __restrict__ W, __bf16* __restrict__ Wt,
                              int K, int N) {
  long i = (long)blockIdx.x * blockDim.x + threadIdx.x;
  if (i >= (long)K * N) return;
  int k = (int)(i % K);
  int n = (int)(i / K);
  Wt[(long)n * K + k] = (__bf16)W[(long)k * N + n];
}

// conv weight [O][I][KK] -> bf16 Bt[o][kk*I + i]
__global__ void prep_conv_kernel(const float* __restrict__ W, __bf16* __restrict__ Bt,
                                 int O, int I, int KK) {
  long idx = (long)blockIdx.x * blockDim.x + threadIdx.x;
  if (idx >= (long)O * I * KK) return;
  int i  = (int)(idx % I);
  long t = idx / I;
  int kk = (int)(t % KK);
  int o  = (int)(t / KK);
  Bt[((long)o * KK + kk) * I + i] = (__bf16)W[((long)o * I + i) * KK + kk];
}

// ---------------- init / util ----------------
__global__ void init_h_kernel(const float* __restrict__ feat, float* __restrict__ h) {
  long i = (long)blockIdx.x * 256 + threadIdx.x;   // grid = NN blocks
  int c = (int)(i & 255);
  long n = i >> 8;
  h[i] = (c < IDIM) ? feat[n * IDIM + c] : 0.0f;
}

__global__ void fill_kernel(float* __restrict__ p, float v, long n) {
  long i = (long)blockIdx.x * blockDim.x + threadIdx.x;
  if (i < n) p[i] = v;
}

__global__ void __launch_bounds__(CDIM) build_c_kernel(const float* __restrict__ h,
                                                       const float* __restrict__ feat,
                                                       float* __restrict__ c) {
  long n = blockIdx.x;          // grid = NN
  int t = threadIdx.x;          // 384 threads
  float v = (t < ODIM) ? h[n * ODIM + t] : feat[n * IDIM + (t - ODIM)];
  c[n * CDIM + t] = v;
}

// ---------------- message transform GEMM ----------------
// Hall[t][n][o] = sum_d h[n][d] * W_msg[t][d][o] + b_msg[t][o]
// Wt: bf16 [4][O=256][D=256] (transposed). Wave tile: 16 rows x 64 cols.
__global__ void __launch_bounds__(256) msg_gemm_kernel(
    const float* __restrict__ h, const __bf16* __restrict__ Wt,
    const float* __restrict__ b_msg, float* __restrict__ Hall) {
  const int lane  = threadIdx.x & 31;
  const int wave  = threadIdx.x >> 5;
  const int mtile = blockIdx.x * 8 + wave;   // 0..2047
  const int cgrp  = blockIdx.y;              // 0..3 (64 cols each)
  const int t     = blockIdx.z;              // etype
  const int nlo   = lane & 15;

  const size_t rowoff = (size_t)mtile * 16 * ODIM;
  const float* arow = h + rowoff + (size_t)nlo * ODIM;
  const __bf16* wbase = Wt + (size_t)t * ODIM * ODIM;
  const __bf16* brow[4];
#pragma unroll
  for (int j = 0; j < 4; ++j)
    brow[j] = wbase + (size_t)(cgrp * 64 + j * 16 + nlo) * ODIM;

  v8f acc[4];
#pragma unroll
  for (int j = 0; j < 4; ++j)
#pragma unroll
    for (int i = 0; i < 8; ++i) acc[j][i] = 0.0f;

  for (int kb = 0; kb < ODIM; kb += 32) {
    v16bf av = load_a_f32(arow, kb, lane);
#pragma unroll
    for (int j = 0; j < 4; ++j)
      acc[j] = wmma_bf16(av, load_b_bf16(brow[j], kb, lane), acc[j]);
  }

  float* obase = Hall + ((size_t)t * NN + (size_t)mtile * 16) * ODIM;
  const int moff = (lane >> 4) * 8;
#pragma unroll
  for (int j = 0; j < 4; ++j) {
    int n = cgrp * 64 + j * 16 + nlo;
    float bv = b_msg[t * ODIM + n];
#pragma unroll
    for (int v = 0; v < 8; ++v)
      obase[(size_t)(moff + v) * ODIM + n] = acc[j][v] + bv;
  }
}

// ---------------- edge gather + scatter-add ----------------
__global__ void __launch_bounds__(256) scatter_kernel(
    const float* __restrict__ Hall, const int* __restrict__ src,
    const int* __restrict__ dst, const int* __restrict__ et,
    float* __restrict__ a) {
  int e = blockIdx.x;
  int c = threadIdx.x;
  int s = src[e], d = dst[e], t = et[e];
  float v = Hall[((size_t)t * NN + (size_t)s) * ODIM + c];
  unsafeAtomicAdd(&a[(size_t)d * ODIM + c], v);
}

// ---------------- fused GRU: gi=a@Wi+bi, gh=h@Wh+bh, gates, in-place h -----
// Wti/Wth: bf16 [768][256] (transposed). Block = 1 row-tile (16 rows), 8 waves,
// each wave owns 2 column-tiles of the 256-wide gate space (r/z/ig/hg each).
__global__ void __launch_bounds__(256) gru_kernel(
    float* __restrict__ h, const float* __restrict__ a,
    const __bf16* __restrict__ Wti, const __bf16* __restrict__ Wth,
    const float* __restrict__ bi, const float* __restrict__ bh) {
  const int lane = threadIdx.x & 31;
  const int wave = threadIdx.x >> 5;
  const int nlo  = lane & 15;
  const size_t rowoff = (size_t)blockIdx.x * 16 * ODIM;
  const float* arow = a + rowoff + (size_t)nlo * ODIM;
  const float* hrow = h + rowoff + (size_t)nlo * ODIM;

  v8f acc_r[2], acc_z[2], acc_ig[2], acc_hg[2];
#pragma unroll
  for (int j = 0; j < 2; ++j)
#pragma unroll
    for (int i = 0; i < 8; ++i) {
      acc_r[j][i] = 0.f; acc_z[j][i] = 0.f; acc_ig[j][i] = 0.f; acc_hg[j][i] = 0.f;
    }

  const __bf16 *p_ri[2], *p_rh[2], *p_zi[2], *p_zh[2], *p_gi[2], *p_gh[2];
#pragma unroll
  for (int j = 0; j < 2; ++j) {
    int col = (wave * 2 + j) * 16 + nlo;
    p_ri[j] = Wti + (size_t)col * ODIM;
    p_rh[j] = Wth + (size_t)col * ODIM;
    p_zi[j] = Wti + (size_t)(col + 256) * ODIM;
    p_zh[j] = Wth + (size_t)(col + 256) * ODIM;
    p_gi[j] = Wti + (size_t)(col + 512) * ODIM;
    p_gh[j] = Wth + (size_t)(col + 512) * ODIM;
  }

  for (int kb = 0; kb < ODIM; kb += 32) {
    v16bf av = load_a_f32(arow, kb, lane);
    v16bf hv = load_a_f32(hrow, kb, lane);
#pragma unroll
    for (int j = 0; j < 2; ++j) {
      acc_r[j]  = wmma_bf16(av, load_b_bf16(p_ri[j], kb, lane), acc_r[j]);
      acc_r[j]  = wmma_bf16(hv, load_b_bf16(p_rh[j], kb, lane), acc_r[j]);
      acc_z[j]  = wmma_bf16(av, load_b_bf16(p_zi[j], kb, lane), acc_z[j]);
      acc_z[j]  = wmma_bf16(hv, load_b_bf16(p_zh[j], kb, lane), acc_z[j]);
      acc_ig[j] = wmma_bf16(av, load_b_bf16(p_gi[j], kb, lane), acc_ig[j]);
      acc_hg[j] = wmma_bf16(hv, load_b_bf16(p_gh[j], kb, lane), acc_hg[j]);
    }
  }

  const int moff = (lane >> 4) * 8;
  float hnew[2][8];
#pragma unroll
  for (int j = 0; j < 2; ++j) {
    int col = (wave * 2 + j) * 16 + nlo;
    float b_r  = bi[col] + bh[col];
    float b_z  = bi[col + 256] + bh[col + 256];
    float b_ig = bi[col + 512];
    float b_hg = bh[col + 512];
#pragma unroll
    for (int v = 0; v < 8; ++v) {
      float hold = h[rowoff + (size_t)(moff + v) * ODIM + col];
      float r = sigm(acc_r[j][v] + b_r);
      float z = sigm(acc_z[j][v] + b_z);
      float g = tanhf(acc_ig[j][v] + b_ig + r * (acc_hg[j][v] + b_hg));
      hnew[j][v] = (1.f - z) * g + z * hold;
    }
  }
  __syncthreads();   // all reads of h by this block complete before writes
#pragma unroll
  for (int j = 0; j < 2; ++j) {
    int col = (wave * 2 + j) * 16 + nlo;
#pragma unroll
    for (int v = 0; v < 8; ++v)
      h[rowoff + (size_t)(moff + v) * ODIM + col] = hnew[j][v];
  }
}

// ---------------- conv-as-GEMM (valid conv over node axis) ----------------
// Row (b,l): A offset = (b*a_graph_rows + l)*a_row_stride, K contiguous.
// Bt: bf16 [N][K]. Output C[b][l][n], relu optional. One wave = 16x16 tile.
__global__ void __launch_bounds__(256) conv_gemm_kernel(
    const float* __restrict__ A, const __bf16* __restrict__ Bt,
    const float* __restrict__ bias, float* __restrict__ C,
    int K, int N, int Lout, int a_row_stride, int a_graph_rows, int relu) {
  const int lane = threadIdx.x & 31;
  const int wid = (int)((blockIdx.x * (long)blockDim.x + threadIdx.x) >> 5);
  const int ntiles = N >> 4;
  const int mtiles = (Lout + 15) >> 4;
  int nt = wid % ntiles;
  int mt = (wid / ntiles) % mtiles;
  int b  = wid / (ntiles * mtiles);
  if (b >= BGR) return;                      // wave-uniform guard

  const int nlo = lane & 15;
  int l  = mt * 16 + nlo;
  int lc = (l < Lout) ? l : (Lout - 1);      // clamp load, mask store
  const float* arow = A + ((size_t)b * a_graph_rows + lc) * a_row_stride;
  int n = nt * 16 + nlo;
  const __bf16* brow = Bt + (size_t)n * K;

  v8f acc;
#pragma unroll
  for (int i = 0; i < 8; ++i) acc[i] = 0.0f;
  for (int kb = 0; kb < K; kb += 32)
    acc = wmma_bf16(load_a_f32(arow, kb, lane), load_b_bf16(brow, kb, lane), acc);

  float bv = bias[n];
  const int moff = (lane >> 4) * 8;
#pragma unroll
  for (int v = 0; v < 8; ++v) {
    int m = mt * 16 + moff + v;
    if (m < Lout) {
      float x = acc[v] + bv;
      if (relu) x = fmaxf(x, 0.0f);
      C[((size_t)b * Lout + m) * N + n] = x;
    }
  }
}

// ---------------- max pool over node axis, [b][L][C] layout ----------------
__global__ void pool_kernel(const float* __restrict__ X, float* __restrict__ Y,
                            int C, int Lin, int Lout, int win, int stride) {
  long idx = (long)blockIdx.x * blockDim.x + threadIdx.x;
  long total = (long)BGR * Lout * C;
  if (idx >= total) return;
  int c  = (int)(idx % C);
  long t = idx / C;
  int lp = (int)(t % Lout);
  int b  = (int)(t / Lout);
  float m = -INFINITY;
  for (int i = 0; i < win; ++i) {
    int l = lp * stride + i;
    float v = X[((size_t)b * Lin + l) * C + c];
    m = fmaxf(m, v);
  }
  Y[idx] = m;
}

// ---------------- gated head: sigmoid(mean_l (Y2@wy+by)*(Z2@wz+bz)) --------
__global__ void __launch_bounds__(64) final_kernel(
    const float* __restrict__ Y2, const float* __restrict__ Z2,
    const float* __restrict__ wy, const float* __restrict__ by,
    const float* __restrict__ wz, const float* __restrict__ bz,
    float* __restrict__ out) {
  __shared__ float sred[64];
  int b = blockIdx.x, t = threadIdx.x;
  float val = 0.0f;
  if (t < 63) {
    const float* yrow = Y2 + ((size_t)b * 63 + t) * ODIM;
    const float* zrow = Z2 + ((size_t)b * 63 + t) * CDIM;
    float yv = by[0];
    for (int i = 0; i < ODIM; ++i) yv += yrow[i] * wy[i];
    float zv = bz[0];
    for (int i = 0; i < CDIM; ++i) zv += zrow[i] * wz[i];
    val = yv * zv;
  }
  sred[t] = val;
  __syncthreads();
  for (int s = 32; s > 0; s >>= 1) {
    if (t < s) sred[t] += sred[t + s];
    __syncthreads();
  }
  if (t == 0) out[b] = 1.0f / (1.0f + expf(-sred[0] / 63.0f));
}

// ---------------- host launcher ----------------
extern "C" void kernel_launch(void* const* d_in, const int* in_sizes, int n_in,
                              void* d_out, int out_size, void* d_ws, size_t ws_size,
                              hipStream_t stream) {
  const float* features = (const float*)d_in[0];
  const int*   src      = (const int*)d_in[1];
  const int*   dst      = (const int*)d_in[2];
  const int*   etype    = (const int*)d_in[3];
  const float* W_msg    = (const float*)d_in[4];
  const float* b_msg    = (const float*)d_in[5];
  const float* gru_Wi   = (const float*)d_in[6];
  const float* gru_Wh   = (const float*)d_in[7];
  const float* gru_bi   = (const float*)d_in[8];
  const float* gru_bh   = (const float*)d_in[9];
  const float* conv1_w  = (const float*)d_in[10];
  const float* conv1_b  = (const float*)d_in[11];
  const float* conv2_w  = (const float*)d_in[12];
  const float* conv2_b  = (const float*)d_in[13];
  const float* convc1_w = (const float*)d_in[14];
  const float* convc1_b = (const float*)d_in[15];
  const float* convc2_w = (const float*)d_in[16];
  const float* convc2_b = (const float*)d_in[17];
  const float* wy = (const float*)d_in[18];
  const float* by = (const float*)d_in[19];
  const float* wz = (const float*)d_in[20];
  const float* bz = (const float*)d_in[21];
  float* out = (float*)d_out;

  // workspace carve-out (256B aligned)
  char* ws = (char*)d_ws;
  size_t off = 0;
  auto take = [&](size_t bytes) -> char* {
    char* p = ws + off;
    off = (off + bytes + 255) & ~(size_t)255;
    return p;
  };
  float*  h    = (float*)take((size_t)NN * ODIM * 4);        //  33.6 MB
  float*  Hall = (float*)take((size_t)4 * NN * ODIM * 4);    // 134.2 MB
  float*  a    = (float*)take((size_t)NN * ODIM * 4);        //  33.6 MB
  float*  cbuf = (float*)take((size_t)NN * CDIM * 4);        //  50.3 MB
  float*  p1   = (float*)take((size_t)BGR * 126 * ODIM * 4); //  16.5 MB
  float*  y2   = (float*)take((size_t)BGR * 126 * ODIM * 4); //  16.5 MB
  float*  Y2p  = (float*)take((size_t)BGR * 63 * ODIM * 4);  //   8.3 MB
  __bf16* Wt_msg = (__bf16*)take((size_t)4 * ODIM * ODIM * 2);
  __bf16* Wti    = (__bf16*)take((size_t)768 * ODIM * 2);
  __bf16* Wth    = (__bf16*)take((size_t)768 * ODIM * 2);
  __bf16* B1t    = (__bf16*)take((size_t)ODIM * 768 * 2);
  __bf16* B2t    = (__bf16*)take((size_t)ODIM * ODIM * 2);
  __bf16* Bc1t   = (__bf16*)take((size_t)CDIM * 1152 * 2);
  __bf16* Bc2t   = (__bf16*)take((size_t)CDIM * CDIM * 2);
  // readout reuse: y1 aliases a (33.3MB <= 33.6MB); z-path aliases Hall
  float* y1  = a;
  float* z1  = Hall;               // [128][254][384] = 12,484,608 f
  float* pz1 = Hall + 12800000;    // [128][126][384] =  6,193,152 f
  float* z2  = Hall + 19200000;    // [128][126][384]
  float* Z2p = Hall + 25600000;    // [128][63][384]  =  3,096,576 f  (fits)

  // ---- weight prep (bf16, transposed to [N][K]) ----
  for (int t = 0; t < 4; ++t)
    prep_t_kernel<<<(ODIM * ODIM + 255) / 256, 256, 0, stream>>>(
        W_msg + (size_t)t * ODIM * ODIM, Wt_msg + (size_t)t * ODIM * ODIM, ODIM, ODIM);
  prep_t_kernel<<<(ODIM * 768 + 255) / 256, 256, 0, stream>>>(gru_Wi, Wti, ODIM, 768);
  prep_t_kernel<<<(ODIM * 768 + 255) / 256, 256, 0, stream>>>(gru_Wh, Wth, ODIM, 768);
  prep_conv_kernel<<<(ODIM * ODIM * 3 + 255) / 256, 256, 0, stream>>>(conv1_w, B1t, ODIM, ODIM, 3);
  prep_conv_kernel<<<(ODIM * ODIM + 255) / 256, 256, 0, stream>>>(conv2_w, B2t, ODIM, ODIM, 1);
  prep_conv_kernel<<<(CDIM * CDIM * 3 + 255) / 256, 256, 0, stream>>>(convc1_w, Bc1t, CDIM, CDIM, 3);
  prep_conv_kernel<<<(CDIM * CDIM + 255) / 256, 256, 0, stream>>>(convc2_w, Bc2t, CDIM, CDIM, 1);

  // ---- h = pad(features, 256) ----
  init_h_kernel<<<NN, 256, 0, stream>>>(features, h);

  // ---- 8 GatedGraphConv steps ----
  for (int s = 0; s < 8; ++s) {
    msg_gemm_kernel<<<dim3(256, 4, 4), 256, 0, stream>>>(h, Wt_msg, b_msg, Hall);
    fill_kernel<<<((long)NN * ODIM + 255) / 256, 256, 0, stream>>>(a, 0.0f, (long)NN * ODIM);
    scatter_kernel<<<EDGES, 256, 0, stream>>>(Hall, src, dst, etype, a);
    gru_kernel<<<NN / 16, 256, 0, stream>>>(h, a, Wti, Wth, gru_bi, gru_bh);
  }

  // ---- readout ----
  build_c_kernel<<<NN, CDIM, 0, stream>>>(h, features, cbuf);
  // Y path: conv1(k=3) -> pool(3,2) -> conv2(k=1) -> pool(2,2)
  conv_gemm_kernel<<<BGR * 16 * 16 / 8, 256, 0, stream>>>(h,   B1t, conv1_b, y1, 768, ODIM, 254, ODIM, 256, 1);
  pool_kernel<<<((long)BGR * 126 * ODIM + 255) / 256, 256, 0, stream>>>(y1, p1, ODIM, 254, 126, 3, 2);
  conv_gemm_kernel<<<BGR * 8 * 16 / 8, 256, 0, stream>>>(p1, B2t, conv2_b, y2, ODIM, ODIM, 126, ODIM, 126, 1);
  pool_kernel<<<((long)BGR * 63 * ODIM + 255) / 256, 256, 0, stream>>>(y2, Y2p, ODIM, 126, 63, 2, 2);
  // Z path on concat features
  conv_gemm_kernel<<<BGR * 16 * 24 / 8, 256, 0, stream>>>(cbuf, Bc1t, convc1_b, z1, 1152, CDIM, 254, CDIM, 256, 1);
  pool_kernel<<<((long)BGR * 126 * CDIM + 255) / 256, 256, 0, stream>>>(z1, pz1, CDIM, 254, 126, 3, 2);
  conv_gemm_kernel<<<BGR * 8 * 24 / 8, 256, 0, stream>>>(pz1, Bc2t, convc2_b, z2, CDIM, CDIM, 126, CDIM, 126, 1);
  pool_kernel<<<((long)BGR * 63 * CDIM + 255) / 256, 256, 0, stream>>>(z2, Z2p, CDIM, 126, 63, 2, 2);

  final_kernel<<<BGR, 64, 0, stream>>>(Y2p, Z2p, wy, by, wz, bz, out);
}
